// GCN_87703232184569
// MI455X (gfx1250) — matrix operand order repaired and checked by reading, not compiled, not run on hardware
//
#include <hip/hip_runtime.h>
#include <hip/hip_bf16.h>

typedef __attribute__((ext_vector_type(16))) _Float16 v16h;
typedef __attribute__((ext_vector_type(8)))  _Float16 v8h;
typedef __attribute__((ext_vector_type(4)))  _Float16 v4h;
typedef __attribute__((ext_vector_type(8)))  float    v8f;

// ---------------------------------------------------------------------------
// Degree / normalization (GCN symmetric norm with self-loops)
// ---------------------------------------------------------------------------
__global__ void k_deg_init(unsigned* __restrict__ deg, int n) {
  int i = blockIdx.x * blockDim.x + threadIdx.x;
  if (i < n) deg[i] = 1u;  // self-loop
}

__global__ void k_deg_edges(const int* __restrict__ ei, unsigned* __restrict__ deg, int e) {
  int i = blockIdx.x * blockDim.x + threadIdx.x;
  if (i < e) atomicAdd(&deg[ei[e + i]], 1u);  // row 1 of edge_index = dst
}

__global__ void k_dinv(const unsigned* __restrict__ deg, float* __restrict__ dinv, int n) {
  int i = blockIdx.x * blockDim.x + threadIdx.x;
  if (i < n) dinv[i] = rsqrtf((float)deg[i]);  // deg >= 1 always
}

// ---------------------------------------------------------------------------
// Aggregation: self-loop term (plain store = init), then edge scatter (atomics)
// cols == 128; float4 per thread; one wave32 covers one row/edge.
// ---------------------------------------------------------------------------
__global__ __launch_bounds__(256)
void k_self_agg(const float* __restrict__ x, const float* __restrict__ dinv,
                float* __restrict__ out, int n, int cols) {
  int t = blockIdx.x * blockDim.x + threadIdx.x;
  int cpt = cols >> 2;
  if (t >= n * cpt) return;
  int i  = t / cpt;
  int c4 = (t % cpt) << 2;
  float s = dinv[i] * dinv[i];
  const float4 v = *(const float4*)(x + (size_t)i * cols + c4);
  float4 r; r.x = v.x * s; r.y = v.y * s; r.z = v.z * s; r.w = v.w * s;
  *(float4*)(out + (size_t)i * cols + c4) = r;
}

__global__ __launch_bounds__(256)
void k_edge_agg(const int* __restrict__ ei, const float* __restrict__ dinv,
                const float* __restrict__ x, float* __restrict__ out,
                int e, int cols) {
  int t = blockIdx.x * blockDim.x + threadIdx.x;
  int cpt = cols >> 2;
  if (t >= e * cpt) return;
  int ed = t / cpt;
  int c4 = (t % cpt) << 2;
  int s = ei[ed];       // src
  int d = ei[e + ed];   // dst
  float nm = dinv[s] * dinv[d];
  const float4 v = *(const float4*)(x + (size_t)s * cols + c4);
  float* ob = out + (size_t)d * cols + c4;
  unsafeAtomicAdd(ob + 0, v.x * nm);
  unsafeAtomicAdd(ob + 1, v.y * nm);
  unsafeAtomicAdd(ob + 2, v.z * nm);
  unsafeAtomicAdd(ob + 3, v.w * nm);
}

// ---------------------------------------------------------------------------
// Bulk f32 -> f16 convert (dims are multiples of 4)
// ---------------------------------------------------------------------------
__global__ __launch_bounds__(256)
void k_cvt_f16(const float* __restrict__ x, _Float16* __restrict__ y, int total4) {
  int t = blockIdx.x * blockDim.x + threadIdx.x;
  if (t >= total4) return;
  const float4 v = *(const float4*)(x + (size_t)t * 4);
  v4h h; h.x = (_Float16)v.x; h.y = (_Float16)v.y; h.z = (_Float16)v.z; h.w = (_Float16)v.w;
  *(v4h*)(y + (size_t)t * 4) = h;
}

// ---------------------------------------------------------------------------
// Pack B[K,N] (f32 row-major) into WMMA-B fragment order, zero padded:
//   Bp[((tn*nChunks + c)*32 + lane)*16 + j]  with  k = c*32 + (lane>>4)*8 + (j&7) + (j>>3)*16,
//                                                 n = tn*16 + (lane&15)
// Each lane's 16 halves for one chunk are 32 contiguous bytes.
// ---------------------------------------------------------------------------
__global__ __launch_bounds__(256)
void k_pack_b(const float* __restrict__ B, _Float16* __restrict__ Bp,
              int K, int N, int Kp, int Np) {
  int t = blockIdx.x * blockDim.x + threadIdx.x;
  int nChunks  = Kp >> 5;
  int perTile  = nChunks << 9;               // nChunks * 512
  int total    = (Np >> 4) * perTile;
  if (t >= total) return;
  int tn = t / perTile;
  int r  = t - tn * perTile;
  int c  = r >> 9;
  int r2 = r & 511;
  int l  = r2 >> 4;
  int j  = r2 & 15;
  int k  = (c << 5) + ((l >> 4) << 3) + (j & 7) + ((j >> 3) << 4);
  int n  = (tn << 4) + (l & 15);
  Bp[t] = (k < K && n < N) ? (_Float16)B[(size_t)k * N + n] : (_Float16)0.0f;
}

// ---------------------------------------------------------------------------
// WMMA GEMM on pre-padded operands: no guards in the K loop.
//   Ah : [M, Kp] f16 row-major (Kp multiple of 32)
//   Bp : fragment-packed f16 (see k_pack_b), padded to Np x Kp with zeros
// Output: either f32 C[M, Nlog] (guarded store) or f16 C[M, Np] (unguarded,
// padding columns receive exact zeros since padded B columns accumulate 0).
// One wave per 16x16 tile; M multiple of 16 (50000 = 3125*16); tile index is
// wave-uniform so EXEC is all ones at every WMMA.
// ---------------------------------------------------------------------------
__global__ __launch_bounds__(256)
void k_wmma_gemm(const _Float16* __restrict__ Ah, const _Float16* __restrict__ Bp,
                 const float* __restrict__ bias,
                 float* __restrict__ Cf, _Float16* __restrict__ Ch,
                 int M, int Np, int Kp, int Nlog, int relu) {
  const int lane = threadIdx.x & 31;
  const int wave = threadIdx.x >> 5;
  const int ntN  = Np >> 4;
  const int ntM  = M >> 4;
  const int tile = blockIdx.x * 8 + wave;
  if (tile >= ntM * ntN) return;
  const int tm = tile / ntN;
  const int tn = tile - tm * ntN;

  const int mA = (tm << 4) + (lane & 15);
  const int nC = (tn << 4) + (lane & 15);
  const int h8 = (lane >> 4) << 3;           // +8 k offset for upper half-wave
  const int nChunks = Kp >> 5;

  const _Float16* __restrict__ arow = Ah + (size_t)mA * Kp + h8;
  const _Float16* __restrict__ bptr = Bp + ((size_t)tn * nChunks << 9) + (lane << 4);

  v8f acc = {};
  for (int c = 0; c < nChunks; ++c) {
    v8h a0 = *(const v8h*)(arow);
    v8h a1 = *(const v8h*)(arow + 16);
    v8h b0 = *(const v8h*)(bptr);
    v8h b1 = *(const v8h*)(bptr + 8);
    v16h a = __builtin_shufflevector(a0, a1, 0,1,2,3,4,5,6,7,8,9,10,11,12,13,14,15);
    v16h b = __builtin_shufflevector(b0, b1, 0,1,2,3,4,5,6,7,8,9,10,11,12,13,14,15);
    acc = __builtin_amdgcn_wmma_f32_16x16x32_f16(
        false, a, false, b, (short)0, acc, false, false);
    arow += 32;
    bptr += 512;
  }

  const bool nok = (nC < Nlog);
  float bv = nok ? bias[nC] : 0.0f;
  if (Ch) {
    // f16 padded output, row stride Np; all lanes store (padding = exact zero)
#pragma unroll
    for (int v = 0; v < 8; ++v) {
      int m = (tm << 4) + v + ((lane >> 4) << 3);
      float val = nok ? (acc[v] + bv) : 0.0f;
      if (relu) val = fmaxf(val, 0.0f);
      Ch[(size_t)m * Np + nC] = (_Float16)val;
    }
  } else if (nok) {
    // f32 output, exact row stride Nlog
#pragma unroll
    for (int v = 0; v < 8; ++v) {
      int m = (tm << 4) + v + ((lane >> 4) << 3);
      float val = acc[v] + bv;
      if (relu) val = fmaxf(val, 0.0f);
      Cf[(size_t)m * Nlog + nC] = val;
    }
  }
}

// ---------------------------------------------------------------------------
// Row softmax, wave32 per row, in place. cols <= 256.
// ---------------------------------------------------------------------------
template <int MAXC>
__global__ __launch_bounds__(256)
void k_softmax(float* __restrict__ p, int n, int cols) {
  int row  = blockIdx.x * 8 + (threadIdx.x >> 5);
  int lane = threadIdx.x & 31;
  if (row >= n) return;
  float* r = p + (size_t)row * cols;
  float vals[MAXC];
  float mx = -3.4e38f;
  int cnt = 0;
  for (int c = lane; c < cols; c += 32) { float v = r[c]; vals[cnt++] = v; mx = fmaxf(mx, v); }
#pragma unroll
  for (int o = 16; o > 0; o >>= 1) mx = fmaxf(mx, __shfl_xor(mx, o, 32));
  float sum = 0.0f;
  cnt = 0;
  for (int c = lane; c < cols; c += 32) { float e = __expf(vals[cnt] - mx); vals[cnt++] = e; sum += e; }
#pragma unroll
  for (int o = 16; o > 0; o >>= 1) sum += __shfl_xor(sum, o, 32);
  float inv = 1.0f / sum;
  cnt = 0;
  for (int c = lane; c < cols; c += 32) r[c] = vals[cnt++] * inv;
}

// ---------------------------------------------------------------------------
// Host-side launch
// ---------------------------------------------------------------------------
static inline int cdiv(long a, long b) { return (int)((a + b - 1) / b); }

extern "C" void kernel_launch(void* const* d_in, const int* in_sizes, int n_in,
                              void* d_out, int out_size, void* d_ws, size_t ws_size,
                              hipStream_t stream) {
  const float* x0  = (const float*)d_in[0];
  const int*   ei  = (const int*)d_in[1];      // [2,E] int32
  const float* W1  = (const float*)d_in[2];  const float* b1  = (const float*)d_in[3];
  const float* Wt  = (const float*)d_in[4];  const float* bt  = (const float*)d_in[5];
  const float* Ws  = (const float*)d_in[6];  const float* bs  = (const float*)d_in[7];
  const float* Wa  = (const float*)d_in[8];  const float* ba  = (const float*)d_in[9];
  const float* Wct = (const float*)d_in[10]; const float* bct = (const float*)d_in[11];
  const float* Wcs = (const float*)d_in[12]; const float* bcs = (const float*)d_in[13];
  const float* Wtf = (const float*)d_in[14]; const float* btf = (const float*)d_in[15];
  const float* Wca = (const float*)d_in[16]; const float* bca = (const float*)d_in[17];

  const int N = in_sizes[0] / 128;   // 50000
  const int E = in_sizes[1] / 2;     // 800000

  // ---- workspace carve (bytes, 256B aligned chunks) ----
  char* base = (char*)d_ws;
  size_t off = 0;
  auto carve = [&](size_t bytes) { void* p = base + off; off = (off + bytes + 255) & ~(size_t)255; return p; };

  float*     dinv = (float*)carve((size_t)N * 4);
  unsigned*  degu = (unsigned*)carve((size_t)N * 4);
  float*     bufA = (float*)carve((size_t)N * 128 * 4);   // aggX then aggH (f32, atomics)
  float*     bufB = (float*)carve((size_t)N * 128 * 4);   // h (f32)
  _Float16*  Ah   = (_Float16*)carve((size_t)N * 128 * 2); // f16 copy of bufA
  _Float16*  x0h  = (_Float16*)carve((size_t)N * 128 * 2);
  _Float16*  ct16 = (_Float16*)carve((size_t)N * 32 * 2);  // conv type  out, Kp=32
  _Float16*  cs16 = (_Float16*)carve((size_t)N * 32 * 2);  // conv school out, Kp=32
  _Float16*  ca16 = (_Float16*)carve((size_t)N * 224 * 2); // conv author out, Kp=224
  // packed B fragments:  size = (Np/16)*(Kp/32)*512 halves
  auto bpSize = [](int Kp, int Np) { return (size_t)(Np >> 4) * (Kp >> 5) * 512; };
  _Float16* BpW1  = (_Float16*)carve(bpSize(128, 128) * 2);
  _Float16* BpWt  = (_Float16*)carve(bpSize(128, 32)  * 2);
  _Float16* BpWs  = (_Float16*)carve(bpSize(128, 32)  * 2);
  _Float16* BpWa  = (_Float16*)carve(bpSize(128, 224) * 2);
  _Float16* BpWct = (_Float16*)carve(bpSize(32, 32)   * 2);
  _Float16* BpWcs = (_Float16*)carve(bpSize(32, 32)   * 2);
  _Float16* BpWca = (_Float16*)carve(bpSize(224, 208) * 2);
  _Float16* BpWtf = (_Float16*)carve(bpSize(128, 16)  * 2);

  // ---- output regions (concatenated flat, return order) ----
  float* out = (float*)d_out;
  float* ot  = out;                   // [N,20]
  float* os  = out + (size_t)N * 20;  // [N,30]
  float* otm = out + (size_t)N * 50;  // [N,15]
  float* oa  = out + (size_t)N * 65;  // [N,200]

  auto packB = [&](const float* B, _Float16* Bp, int K, int Ncols, int Kp, int Np) {
    int total = (int)bpSize(Kp, Np);
    k_pack_b<<<cdiv(total, 256), 256, 0, stream>>>(B, Bp, K, Ncols, Kp, Np);
  };
  auto gemm = [&](const _Float16* A, const _Float16* Bp, const float* bias,
                  float* Cf, _Float16* Ch, int Np, int Kp, int Nlog, int relu) {
    long tiles = (long)(N >> 4) * (Np >> 4);
    k_wmma_gemm<<<cdiv(tiles, 8), 256, 0, stream>>>(A, Bp, bias, Cf, Ch, N, Np, Kp, Nlog, relu);
  };

  // 0) pack all weights into WMMA fragment order (tiny)
  packB(W1,  BpW1,  128, 128, 128, 128);
  packB(Wt,  BpWt,  128, 20,  128, 32);
  packB(Ws,  BpWs,  128, 30,  128, 32);
  packB(Wa,  BpWa,  128, 200, 128, 224);
  packB(Wct, BpWct, 20,  20,  32,  32);
  packB(Wcs, BpWcs, 30,  30,  32,  32);
  packB(Wca, BpWca, 200, 200, 224, 208);
  packB(Wtf, BpWtf, 128, 15,  128, 16);
  k_cvt_f16<<<cdiv((long)N * 32, 256), 256, 0, stream>>>(x0, x0h, N * 32);

  // 1) degree + symmetric normalization
  k_deg_init<<<cdiv(N, 256), 256, 0, stream>>>(degu, N);
  k_deg_edges<<<cdiv(E, 256), 256, 0, stream>>>(ei, degu, E);
  k_dinv<<<cdiv(N, 256), 256, 0, stream>>>(degu, dinv, N);

  // 2) aggX = D^-1/2 (A+I) D^-1/2 x0
  k_self_agg<<<cdiv((long)N * 32, 256), 256, 0, stream>>>(x0, dinv, bufA, N, 128);
  k_edge_agg<<<cdiv((long)E * 32, 256), 256, 0, stream>>>(ei, dinv, x0, bufA, E, 128);
  k_cvt_f16<<<cdiv((long)N * 32, 256), 256, 0, stream>>>(bufA, Ah, N * 32);

  // 3) h = relu(aggX @ W1 + b1)   (f32 out, feeds fp32 atomic aggregation)
  gemm(Ah, BpW1, b1, bufB, nullptr, 128, 128, 128, 1);

  // 4) aggH = D^-1/2 (A+I) D^-1/2 h
  k_self_agg<<<cdiv((long)N * 32, 256), 256, 0, stream>>>(bufB, dinv, bufA, N, 128);
  k_edge_agg<<<cdiv((long)E * 32, 256), 256, 0, stream>>>(ei, dinv, bufB, bufA, E, 128);
  k_cvt_f16<<<cdiv((long)N * 32, 256), 256, 0, stream>>>(bufA, Ah, N * 32);

  // 5) conv outputs (f16 padded, feed head GEMMs directly)
  gemm(Ah, BpWt, bt, nullptr, ct16, 32,  128, 20,  0);
  gemm(Ah, BpWs, bs, nullptr, cs16, 32,  128, 30,  0);
  gemm(Ah, BpWa, ba, nullptr, ca16, 224, 128, 200, 0);

  // 6) heads -> d_out (f32, exact strides)
  gemm(ct16, BpWct, bct, ot,  nullptr, 32,  32,  20,  0);
  gemm(cs16, BpWcs, bcs, os,  nullptr, 32,  32,  30,  0);
  gemm(x0h,  BpWtf, btf, otm, nullptr, 16,  128, 15,  0);
  gemm(ca16, BpWca, bca, oa,  nullptr, 208, 224, 200, 0);

  // 7) in-place row softmax
  k_softmax<8><<<cdiv(N, 8), 256, 0, stream>>>(ot,  N, 20);
  k_softmax<8><<<cdiv(N, 8), 256, 0, stream>>>(os,  N, 30);
  k_softmax<8><<<cdiv(N, 8), 256, 0, stream>>>(otm, N, 15);
  k_softmax<8><<<cdiv(N, 8), 256, 0, stream>>>(oa,  N, 200);
}
